// Experts_62508954026515
// MI455X (gfx1250) — matrix operand order repaired and checked by reading, not compiled, not run on hardware
//
#include <hip/hip_runtime.h>

namespace {

constexpr int kE = 8;
constexpr int kC = 1024;
constexpr int kH = 2048;
constexpr int kI = 5632;

constexpr int BK   = 64;   // K-step per LDS stage (2 WMMA K-steps)
constexpr int LDSS = 72;   // padded LDS row stride in bf16 (144B; 36*r mod 64 distinct)

typedef __attribute__((ext_vector_type(16))) __bf16 v16bf;
typedef __attribute__((ext_vector_type(8)))  __bf16 bf16x8;
typedef __attribute__((ext_vector_type(4)))  __bf16 bf16x4;
typedef __attribute__((ext_vector_type(2)))  __bf16 bf16x2;
typedef __attribute__((ext_vector_type(8)))  float  v8f;

union FragBF16 {
  bf16x8 h[2];
  v16bf  v;
};

__device__ __forceinline__ __bf16 f32_to_bf16(float f) {
  unsigned u = __builtin_bit_cast(unsigned, f);
  u += 0x7FFFu + ((u >> 16) & 1u);          // round-to-nearest-even
  unsigned short hs = (unsigned short)(u >> 16);
  return __builtin_bit_cast(__bf16, hs);
}

// Packed f32x2 -> bf16x2.
//  1) native v_cvt_pk_bf16_f32 if the builtin exists (it did not last round)
//  2) round-to-nearest + single v_perm_b32 pack: 3 VALU / 2 elements
//  3) scalar RNE fallback (host pass)
__device__ __forceinline__ bf16x2 cvt2(float a, float b) {
#if __has_builtin(__builtin_amdgcn_cvt_pk_bf16_f32)
  return __builtin_amdgcn_cvt_pk_bf16_f32(a, b);
#elif __has_builtin(__builtin_amdgcn_perm)
  const unsigned ua = __builtin_bit_cast(unsigned, a) + 0x8000u;
  const unsigned ub = __builtin_bit_cast(unsigned, b) + 0x8000u;
  const unsigned p  = __builtin_amdgcn_perm(ub, ua, 0x07060302u);
  return __builtin_bit_cast(bf16x2, p);
#else
  bf16x2 r;
  r[0] = f32_to_bf16(a);
  r[1] = f32_to_bf16(b);
  return r;
#endif
}

__device__ __forceinline__ bf16x8 cvt8(const float4 a, const float4 b) {
  union { bf16x2 p[4]; bf16x8 v; } u;
  u.p[0] = cvt2(a.x, a.y);
  u.p[1] = cvt2(a.z, a.w);
  u.p[2] = cvt2(b.x, b.y);
  u.p[3] = cvt2(b.z, b.w);
  return u.v;
}

// 64B per lane: global -> LDS via CDNA5 async DMA (ASYNCcnt tracked).
// ISA 10.x: INST_OFFSET is added to BOTH the LDS and global addresses.
__device__ __forceinline__ void async_copy64(unsigned ldsDst, unsigned vOff,
                                             const void* sBase) {
  asm volatile(
      "global_load_async_to_lds_b128 %0, %1, %2 offset:0\n\t"
      "global_load_async_to_lds_b128 %0, %1, %2 offset:16\n\t"
      "global_load_async_to_lds_b128 %0, %1, %2 offset:32\n\t"
      "global_load_async_to_lds_b128 %0, %1, %2 offset:48"
      :: "v"(ldsDst), "v"(vOff), "s"(sBase) : "memory");
}

__device__ __forceinline__ void wait_async0() {
#if __has_builtin(__builtin_amdgcn_s_wait_asynccnt)
  __builtin_amdgcn_s_wait_asynccnt(0);
#else
  asm volatile("s_wait_asynccnt 0" ::: "memory");
#endif
}

// ---------------------------------------------------------------------------
// Pre-pass: X f32 -> bf16 once (16.8M elems, ~3us at 23.3TB/s).
// ---------------------------------------------------------------------------
__global__ __launch_bounds__(256) void cvt_bf16_kernel(
    const float* __restrict__ in, __bf16* __restrict__ outb)
{
  const size_t i = (size_t)blockIdx.x * 256 + threadIdx.x;
  const float4 v = ((const float4*)in)[i];
  union { bf16x2 p[2]; bf16x4 v4; } u;
  u.p[0] = cvt2(v.x, v.y);
  u.p[1] = cvt2(v.z, v.w);
  *(bf16x4*)(outb + i * 4) = u.v4;
}

// ---------------------------------------------------------------------------
// Kernel A: inter[e,c,i] = silu(X·Gt) * (X·Ut)   (bf16 into workspace)
// Block tile 128(C) x 64(I), K-stage 64 over H. 8 waves (4x2), 16 WMMA/wave/it.
// A tile: async DMA bf16 -> double-buffered LDS. G/U: reg-staged f32 -> cvt.
// ---------------------------------------------------------------------------
__global__ __launch_bounds__(256) void gateup_kernel(
    const __bf16* __restrict__ xb, const float* __restrict__ gw,
    const float* __restrict__ uw, __bf16* __restrict__ inter)
{
  __shared__ alignas(16) __bf16 sA[2][128 * LDSS];
  __shared__ alignas(16) __bf16 sG[64 * LDSS];
  __shared__ alignas(16) __bf16 sU[64 * LDSS];

  const int e  = blockIdx.z;
  const int c0 = blockIdx.y * 128;
  const int i0 = blockIdx.x * 64;

  const int t     = threadIdx.x;
  const int lane  = t & 31;
  const int wid   = t >> 5;
  const int waveM = wid & 3;     // 32 rows each
  const int waveN = wid >> 2;    // 32 cols each
  const int hseg  = lane >> 4;   // 0/1 (K-half select)
  const int l16   = lane & 15;

  const __bf16* xBase = xb + ((size_t)e * kC + c0) * kH;
  const float*  gBase = gw + ((size_t)e * kI + i0) * kH;
  const float*  uBase = uw + ((size_t)e * kI + i0) * kH;

  const int aRow = t >> 1, aCol = (t & 1) * 32;   // 32 bf16 / thread (64B)
  const int gRow = t >> 2, gCol = (t & 3) * 16;   // 16 floats / thread

  const unsigned sA0  = (unsigned)(uintptr_t)(&sA[0][0]);
  const unsigned aDst = sA0 + (unsigned)(aRow * (LDSS * 2) + aCol * 2);

  float4 gReg[4], uReg[4];
  auto loadGU = [&](int k) {
    const float* gp = gBase + (size_t)gRow * kH + k + gCol;
    gReg[0] = *(const float4*)(gp + 0);
    gReg[1] = *(const float4*)(gp + 4);
    gReg[2] = *(const float4*)(gp + 8);
    gReg[3] = *(const float4*)(gp + 12);
    const float* up = uBase + (size_t)gRow * kH + k + gCol;
    uReg[0] = *(const float4*)(up + 0);
    uReg[1] = *(const float4*)(up + 4);
    uReg[2] = *(const float4*)(up + 8);
    uReg[3] = *(const float4*)(up + 12);
  };
  auto asyncA = [&](int k, int buf) {
    const unsigned vOff = (unsigned)((aRow * kH + k + aCol) * 2);
    async_copy64(aDst + (unsigned)(buf * 128 * LDSS * 2), vOff, (const void*)xBase);
  };

  v8f accG[2][2] = {};
  v8f accU[2][2] = {};

  asyncA(0, 0);
  loadGU(0);
  int cur = 0;
  for (int k = 0; k < kH; k += BK) {
    __syncthreads();                                  // prior compute done
    *(bf16x8*)(&sG[gRow * LDSS + gCol])     = cvt8(gReg[0], gReg[1]);
    *(bf16x8*)(&sG[gRow * LDSS + gCol + 8]) = cvt8(gReg[2], gReg[3]);
    *(bf16x8*)(&sU[gRow * LDSS + gCol])     = cvt8(uReg[0], uReg[1]);
    *(bf16x8*)(&sU[gRow * LDSS + gCol + 8]) = cvt8(uReg[2], uReg[3]);
    wait_async0();                                    // A DMA for buf[cur] done
    __syncthreads();                                  // tiles visible to all
    if (k + BK < kH) { asyncA(k + BK, cur ^ 1); loadGU(k + BK); }

#pragma unroll
    for (int kk = 0; kk < BK; kk += 32) {
      FragBF16 a[2], g[2], u[2];
#pragma unroll
      for (int mt = 0; mt < 2; ++mt) {
        const __bf16* pa =
            &sA[cur][(waveM * 32 + mt * 16 + l16) * LDSS + kk + hseg * 16];
        a[mt].h[0] = *(const bf16x8*)(pa);
        a[mt].h[1] = *(const bf16x8*)(pa + 8);
      }
#pragma unroll
      for (int nt = 0; nt < 2; ++nt) {
        const __bf16* pg =
            &sG[(waveN * 32 + nt * 16 + l16) * LDSS + kk + hseg * 16];
        g[nt].h[0] = *(const bf16x8*)(pg);
        g[nt].h[1] = *(const bf16x8*)(pg + 8);
        const __bf16* pu =
            &sU[(waveN * 32 + nt * 16 + l16) * LDSS + kk + hseg * 16];
        u[nt].h[0] = *(const bf16x8*)(pu);
        u[nt].h[1] = *(const bf16x8*)(pu + 8);
      }
#pragma unroll
      for (int mt = 0; mt < 2; ++mt)
#pragma unroll
        for (int nt = 0; nt < 2; ++nt) {
          accG[mt][nt] = __builtin_amdgcn_wmma_f32_16x16x32_bf16(
              false, a[mt].v, false, g[nt].v, (short)0, accG[mt][nt], false, false);
          accU[mt][nt] = __builtin_amdgcn_wmma_f32_16x16x32_bf16(
              false, a[mt].v, false, u[nt].v, (short)0, accU[mt][nt], false, false);
        }
    }
    cur ^= 1;
  }

  // epilogue: silu(gate)*up -> bf16 inter (packed conversion in pairs)
#pragma unroll
  for (int mt = 0; mt < 2; ++mt)
#pragma unroll
    for (int nt = 0; nt < 2; ++nt) {
      const int col = i0 + waveN * 32 + nt * 16 + l16;
      float sv[8];
#pragma unroll
      for (int r = 0; r < 8; ++r) {
        const float gv = accG[mt][nt][r];
        sv[r] = (gv / (1.0f + __expf(-gv))) * accU[mt][nt][r];
      }
#pragma unroll
      for (int r = 0; r < 8; r += 2) {
        const bf16x2 p = cvt2(sv[r], sv[r + 1]);
        const int row = c0 + waveM * 32 + mt * 16 + hseg * 8 + r;
        inter[((size_t)e * kC + row) * kI + col]     = p[0];
        inter[((size_t)e * kC + row + 1) * kI + col] = p[1];
      }
    }
}

// ---------------------------------------------------------------------------
// Kernel B: out[e,c,h] = inter[e,c,:] · down[e,h,:]
// Block tile 128(C) x 128(H), K-stage 64 over I. 8 waves (4x2, 32x64 each).
// A tile (inter, bf16): async DMA -> double-buffered LDS. B: reg-staged cvt.
// ---------------------------------------------------------------------------
__global__ __launch_bounds__(256) void down_kernel(
    const __bf16* __restrict__ inter, const float* __restrict__ dw,
    float* __restrict__ out)
{
  __shared__ alignas(16) __bf16 sA[2][128 * LDSS];
  __shared__ alignas(16) __bf16 sB[128 * LDSS];

  const int e  = blockIdx.z;
  const int c0 = blockIdx.y * 128;
  const int h0 = blockIdx.x * 128;

  const int t     = threadIdx.x;
  const int lane  = t & 31;
  const int wid   = t >> 5;
  const int waveM = wid & 3;     // 32 rows
  const int waveN = wid >> 2;    // 64 cols
  const int hseg  = lane >> 4;
  const int l16   = lane & 15;

  const __bf16* iBase = inter + ((size_t)e * kC + c0) * kI;
  const float*  dBase = dw    + ((size_t)e * kH + h0) * kI;

  const int aRow = t >> 1, aCol = (t & 1) * 32;   // 32 bf16 / thread (64B)
  const int bRow = t >> 1, bCol = (t & 1) * 32;   // 32 floats / thread

  const unsigned sA0  = (unsigned)(uintptr_t)(&sA[0][0]);
  const unsigned aDst = sA0 + (unsigned)(aRow * (LDSS * 2) + aCol * 2);

  float4 bS[8];
  auto loadB = [&](int k) {
    const float* bp = dBase + (size_t)bRow * kI + k + bCol;
#pragma unroll
    for (int j = 0; j < 8; ++j) bS[j] = *(const float4*)(bp + 4 * j);
  };
  auto asyncA = [&](int k, int buf) {
    const unsigned vOff = (unsigned)((aRow * kI + k + aCol) * 2);
    async_copy64(aDst + (unsigned)(buf * 128 * LDSS * 2), vOff, (const void*)iBase);
  };

  v8f acc[2][4] = {};

  asyncA(0, 0);
  loadB(0);
  int cur = 0;
  for (int k = 0; k < kI; k += BK) {
    __syncthreads();
    *(bf16x8*)(&sB[bRow * LDSS + bCol])      = cvt8(bS[0], bS[1]);
    *(bf16x8*)(&sB[bRow * LDSS + bCol + 8])  = cvt8(bS[2], bS[3]);
    *(bf16x8*)(&sB[bRow * LDSS + bCol + 16]) = cvt8(bS[4], bS[5]);
    *(bf16x8*)(&sB[bRow * LDSS + bCol + 24]) = cvt8(bS[6], bS[7]);
    wait_async0();
    __syncthreads();
    if (k + BK < kI) { asyncA(k + BK, cur ^ 1); loadB(k + BK); }

#pragma unroll
    for (int kk = 0; kk < BK; kk += 32) {
      FragBF16 a[2], b[4];
#pragma unroll
      for (int mt = 0; mt < 2; ++mt) {
        const __bf16* pa =
            &sA[cur][(waveM * 32 + mt * 16 + l16) * LDSS + kk + hseg * 16];
        a[mt].h[0] = *(const bf16x8*)(pa);
        a[mt].h[1] = *(const bf16x8*)(pa + 8);
      }
#pragma unroll
      for (int nt = 0; nt < 4; ++nt) {
        const __bf16* pb =
            &sB[(waveN * 64 + nt * 16 + l16) * LDSS + kk + hseg * 16];
        b[nt].h[0] = *(const bf16x8*)(pb);
        b[nt].h[1] = *(const bf16x8*)(pb + 8);
      }
#pragma unroll
      for (int mt = 0; mt < 2; ++mt)
#pragma unroll
        for (int nt = 0; nt < 4; ++nt)
          acc[mt][nt] = __builtin_amdgcn_wmma_f32_16x16x32_bf16(
              false, a[mt].v, false, b[nt].v, (short)0, acc[mt][nt], false, false);
    }
    cur ^= 1;
  }

#pragma unroll
  for (int mt = 0; mt < 2; ++mt)
#pragma unroll
    for (int nt = 0; nt < 4; ++nt) {
      const int col = h0 + waveN * 64 + nt * 16 + l16;
#pragma unroll
      for (int r = 0; r < 8; ++r) {
        const int row = c0 + waveM * 32 + mt * 16 + hseg * 8 + r;
        out[((size_t)e * kC + row) * kH + col] = acc[mt][nt][r];
      }
    }
}

} // namespace

extern "C" void kernel_launch(void* const* d_in, const int* in_sizes, int n_in,
                              void* d_out, int out_size, void* d_ws, size_t ws_size,
                              hipStream_t stream) {
  (void)in_sizes; (void)n_in; (void)out_size; (void)ws_size;
  const float* x  = (const float*)d_in[0];   // [E,C,H] f32
  const float* gw = (const float*)d_in[1];   // [E,I,H] f32
  const float* uw = (const float*)d_in[2];   // [E,I,H] f32
  const float* dw = (const float*)d_in[3];   // [E,H,I] f32
  float* out = (float*)d_out;                // [E,C,H] f32

  // workspace layout: xbf [E,C,H] bf16 (33.6MB) | inter [E,C,I] bf16 (92.3MB)
  __bf16* xbf   = (__bf16*)d_ws;
  __bf16* inter = (__bf16*)((char*)d_ws + (size_t)kE * kC * kH * sizeof(__bf16));

  cvt_bf16_kernel<<<kE * kC * kH / 4 / 256, 256, 0, stream>>>(x, xbf);

  dim3 gridA(kI / 64, kC / 128, kE);   // 88 x 8 x 8
  gateup_kernel<<<gridA, 256, 0, stream>>>(xbf, gw, uw, inter);

  dim3 gridB(kH / 128, kC / 128, kE);  // 16 x 8 x 8
  down_kernel<<<gridB, 256, 0, stream>>>(inter, dw, out);
}
